// SpikesToAnalog_68401649156432
// MI455X (gfx1250) — compile-verified, matrix operand order checked by reading, not run on hardware
//
#include <hip/hip_runtime.h>
#include <stdint.h>

typedef __attribute__((ext_vector_type(16))) _Float16 v16h;
typedef __attribute__((ext_vector_type(8)))  _Float16 v8h;
typedef __attribute__((ext_vector_type(4)))  _Float16 v4h;
typedef __attribute__((ext_vector_type(8)))  float    v8f;
typedef __attribute__((ext_vector_type(4)))  float    v4f;

#define TT    32
#define DIN   1024
#define DOUT  1024
#define NROWS 4096                  // B*S = 8*512
#define DECAY 0.95122942450071400f  // exp(-1/20)

// ---------------------------------------------------------------------------
// Kernel 1: leaky integration over T (Horner: acc = acc*decay + x_t).
// 512 MiB HBM stream -> ~22 us roofline at 23.3 TB/s. Loads are non-temporal
// so the once-read x stream doesn't evict mem/W from the 192 MB L2.
// ---------------------------------------------------------------------------
__global__ __launch_bounds__(256) void decay_reduce_f16(
    const float* __restrict__ x, _Float16* __restrict__ memh) {
  int idx = blockIdx.x * blockDim.x + threadIdx.x;   // one per 4 elements
  int n  = idx >> 8;          // row (0..4095)
  int d4 = (idx & 255) << 2;  // d = d4..d4+3
  const float* p = x + (size_t)n * (TT * DIN) + d4;
  float a0 = 0.f, a1 = 0.f, a2 = 0.f, a3 = 0.f;
#pragma unroll
  for (int t = 0; t < TT; ++t) {
    v4f v = __builtin_nontemporal_load((const v4f*)(p + (size_t)t * DIN));
    a0 = a0 * DECAY + v.x;
    a1 = a1 * DECAY + v.y;
    a2 = a2 * DECAY + v.z;
    a3 = a3 * DECAY + v.w;
  }
  v4h h;
  h.x = (_Float16)a0; h.y = (_Float16)a1;
  h.z = (_Float16)a2; h.w = (_Float16)a3;
  *(v4h*)(memh + (size_t)n * DIN + d4) = h;
}

// ---------------------------------------------------------------------------
// Kernel 2: W fp32 -> f16 (row-major [DOUT, DIN]); W_f16 stays L2-resident.
// ---------------------------------------------------------------------------
__global__ __launch_bounds__(256) void convert_w_f16(
    const float* __restrict__ W, _Float16* __restrict__ Wh) {
  int idx = (blockIdx.x * blockDim.x + threadIdx.x) << 2;
  v4f w = *(const v4f*)(W + idx);
  v4h h;
  h.x = (_Float16)w.x; h.y = (_Float16)w.y;
  h.z = (_Float16)w.z; h.w = (_Float16)w.w;
  *(v4h*)(Wh + idx) = h;
}

// ---------------------------------------------------------------------------
// CDNA5 async copy: global -> LDS, 16 bytes per lane (ASYNCcnt-tracked).
// ---------------------------------------------------------------------------
__device__ __forceinline__ void async_ld_b128(uint32_t lds_off,
                                              const _Float16* g) {
  asm volatile("global_load_async_to_lds_b128 %0, %1, off"
               :: "v"(lds_off), "v"(g) : "memory");
}
__device__ __forceinline__ void wait_async_all() {
  asm volatile("s_wait_asynccnt 0x0" ::: "memory");
}

__device__ __forceinline__ v16h combine(v8h lo, v8h hi) {
  v16h r;
#pragma unroll
  for (int i = 0; i < 8; ++i) { r[i] = lo[i]; r[i + 8] = hi[i]; }
  return r;
}

// ---------------------------------------------------------------------------
// Kernel 3: out[m,n] = sum_k mem[m,k] * W[n,k] + bias[n]
//
// Block = 128(M) x 128(N), 8 waves as 2(M) x 4(N); wave = 64(M) x 32(N)
// -> 8 WMMA accumulators. A K-chunks double-buffered in LDS via
// global_load_async_to_lds_b128; B fragments register-ping-ponged one K-step
// ahead. K-loop manually unrolled x2 with alternating buffer/register sets so
// no register rotation (v_mov chains) is needed between steps.
// ---------------------------------------------------------------------------
#define GEMM_STEP(KIDX, CUR, BC0, BC1, BN0, BN1, PREFETCH)                     \
  {                                                                            \
    if (PREFETCH) {                                                            \
      const int k1 = ((KIDX) + 1) * 32;                                        \
      uint32_t l0 = (uint32_t)(uintptr_t)&shA[(CUR) ^ 1][segR][segC];          \
      async_ld_b128(l0, aBase + (size_t)segR * DIN + k1 + segC);               \
      uint32_t l1 = (uint32_t)(uintptr_t)&shA[(CUR) ^ 1][segR + 64][segC];     \
      async_ld_b128(l1, aBase + (size_t)(segR + 64) * DIN + k1 + segC);        \
      BN0 = combine(*(const v8h*)(bRow0 + k1), *(const v8h*)(bRow0 + k1 + 16));\
      BN1 = combine(*(const v8h*)(bRow1 + k1), *(const v8h*)(bRow1 + k1 + 16));\
    }                                                                          \
    v16h af[4];                                                                \
    _Pragma("unroll")                                                          \
    for (int m = 0; m < 4; ++m) {                                              \
      const _Float16* ap = &shA[CUR][waveM * 64 + m * 16 + ln][koff];          \
      af[m] = combine(*(const v8h*)ap, *(const v8h*)(ap + 16));                \
    }                                                                          \
    _Pragma("unroll")                                                          \
    for (int m = 0; m < 4; ++m) {                                              \
      acc[m][0] = __builtin_amdgcn_wmma_f32_16x16x32_f16(                      \
          false, af[m], false, BC0, (short)0, acc[m][0], false, false);        \
      acc[m][1] = __builtin_amdgcn_wmma_f32_16x16x32_f16(                      \
          false, af[m], false, BC1, (short)0, acc[m][1], false, false);        \
    }                                                                          \
    wait_async_all();                                                          \
    __syncthreads();                                                           \
  }

__global__ __launch_bounds__(256) void wmma_gemm_f16(
    const _Float16* __restrict__ A,   // mem_f16 [NROWS, DIN]
    const _Float16* __restrict__ Bw,  // W_f16   [DOUT, DIN]
    const float* __restrict__ bias,   // [DOUT]
    float* __restrict__ out) {        // [NROWS, DOUT]
  __shared__ _Float16 shA[2][128][32];   // 16 KB, double-buffered A chunk

  const int tid   = threadIdx.x;
  const int lane  = tid & 31;
  const int wave  = tid >> 5;
  const int bm    = blockIdx.x >> 3;     // 0..31  (M / 128)
  const int bn    = blockIdx.x & 7;      // 0..7   (N / 128)
  const int waveM = wave >> 2;           // 0..1
  const int waveN = wave & 3;            // 0..3
  const int ln    = lane & 15;
  const int hi    = lane >> 4;
  const int koff  = hi << 3;             // 0 or 8 halves

  // async staging: each thread owns two 16B segments of the 128x32 A chunk
  const int segR = tid >> 2;             // 0..63
  const int segC = (tid & 3) << 3;       // 0,8,16,24 (halves)
  const _Float16* aBase = A + (size_t)(bm * 128) * DIN;

  // B fragment source rows (W rows = output cols)
  const int nBase = bn * 128 + waveN * 32;
  const _Float16* bRow0 = Bw + (size_t)(nBase + ln) * DIN + koff;
  const _Float16* bRow1 = bRow0 + (size_t)16 * DIN;

  v8f acc[4][2];
#pragma unroll
  for (int m = 0; m < 4; ++m)
#pragma unroll
    for (int n = 0; n < 2; ++n) acc[m][n] = (v8f){};

  // prologue: stage K-chunk 0 into LDS buf 0; load B fragments for k=0
  {
    uint32_t l0 = (uint32_t)(uintptr_t)&shA[0][segR][segC];
    async_ld_b128(l0, aBase + (size_t)segR * DIN + segC);
    uint32_t l1 = (uint32_t)(uintptr_t)&shA[0][segR + 64][segC];
    async_ld_b128(l1, aBase + (size_t)(segR + 64) * DIN + segC);
  }
  v16h bcA0 = combine(*(const v8h*)(bRow0), *(const v8h*)(bRow0 + 16));
  v16h bcA1 = combine(*(const v8h*)(bRow1), *(const v8h*)(bRow1 + 16));
  v16h bcB0, bcB1;
  wait_async_all();
  __syncthreads();

#pragma unroll 1
  for (int k = 0; k < DIN / 32; k += 2) {
    // even step: consume LDS buf 0 + bcA, stage buf 1 + bcB
    GEMM_STEP(k,     0, bcA0, bcA1, bcB0, bcB1, true);
    // odd step: consume LDS buf 1 + bcB, stage buf 0 + bcA (skip on last)
    GEMM_STEP(k + 1, 1, bcB0, bcB1, bcA0, bcA1, (k + 1 < DIN / 32 - 1));
  }

  // epilogue: C/D layout -> lane<16: M=r, lane>=16: M=r+8; N=lane&15
#pragma unroll
  for (int n = 0; n < 2; ++n) {
    const int col = nBase + n * 16 + ln;
    const float bcv = bias[col];
#pragma unroll
    for (int m = 0; m < 4; ++m) {
      float* o = out +
          (size_t)(bm * 128 + waveM * 64 + m * 16 + (hi << 3)) * DOUT + col;
#pragma unroll
      for (int r = 0; r < 8; ++r)
        __builtin_nontemporal_store(acc[m][n][r] + bcv, o + (size_t)r * DOUT);
    }
  }
}

// ---------------------------------------------------------------------------
extern "C" void kernel_launch(void* const* d_in, const int* in_sizes, int n_in,
                              void* d_out, int out_size, void* d_ws, size_t ws_size,
                              hipStream_t stream) {
  const float* x  = (const float*)d_in[0];  // [8,512,32,1024]
  const float* W  = (const float*)d_in[1];  // [1024,1024]
  const float* b  = (const float*)d_in[2];  // [1024]
  float* out = (float*)d_out;               // [8,512,1024]

  _Float16* memh = (_Float16*)d_ws;                 // 8 MiB
  _Float16* Wh   = memh + (size_t)NROWS * DIN;      // +2 MiB

  decay_reduce_f16<<<(NROWS * DIN / 4) / 256, 256, 0, stream>>>(x, memh);
  convert_w_f16<<<(DOUT * DIN / 4) / 256, 256, 0, stream>>>(W, Wh);
  wmma_gemm_f16<<<(NROWS / 128) * (DOUT / 128), 256, 0, stream>>>(memh, Wh, b, out);
}
